// BahdanauAttention_12275016532424
// MI455X (gfx1250) — matrix operand order repaired and checked by reading, compiled, not run on hardware
//
#include <hip/hip_runtime.h>
#include <hip/hip_bf16.h>
#include <math.h>

#define BB  32
#define SS  8192
#define HEc 256
#define HDc 256
#define WVS 4      // waves per block in the scores kernel
#define SPW 8      // strips (16 rows of S) per wave

typedef __attribute__((ext_vector_type(16))) __bf16 v16bf;
typedef __attribute__((ext_vector_type(8)))  __bf16 v8bf;
typedef __attribute__((ext_vector_type(8)))  float  v8f;
typedef __attribute__((ext_vector_type(4)))  float  vf4;

// Branchless tanh: hardware v_tanh_f32 on gfx1250 if the toolchain exposes it,
// else v_exp_f32 + v_rcp_f32 (saturates correctly at +-1, no EXEC divergence).
static __device__ __forceinline__ float fast_tanh(float x) {
#if __has_builtin(__builtin_amdgcn_tanhf)
    return __builtin_amdgcn_tanhf(x);
#else
    const float e = __expf(x + x);                       // v_exp_f32 path
    return 1.0f - 2.0f * __builtin_amdgcn_rcpf(e + 1.0f);
#endif
}

// ---------------------------------------------------------------------------
// Kernel 1: prep. Convert Ua_w -> bf16 and fold all biases + q-projection into
// qpb[b][h] = sum_d query[b,d]*Wa_w[h,d] + Wa_b[h] + Ua_b[h]
// grid: 32 blocks (b), 256 threads (h)
// ---------------------------------------------------------------------------
__global__ __launch_bounds__(256)
void bah_prep(const float* __restrict__ query, const float* __restrict__ Wa_w,
              const float* __restrict__ Wa_b,  const float* __restrict__ Ua_w,
              const float* __restrict__ Ua_b,
              __bf16* __restrict__ ua_bf, float* __restrict__ qpb) {
    const int b = blockIdx.x;
    const int h = threadIdx.x;
    #pragma unroll
    for (int r = 0; r < 8; ++r) {
        const int row = b * 8 + r;
        ua_bf[row * HEc + h] = (__bf16)Ua_w[row * HEc + h];
    }
    float acc = Wa_b[h] + Ua_b[h];
    const float* q = query + b * HDc;
    const float* w = Wa_w + h * HDc;
    #pragma unroll 4
    for (int d = 0; d < HDc; ++d) acc = fmaf(q[d], w[d], acc);
    qpb[b * HDc + h] = acc;
}

// ---------------------------------------------------------------------------
// Kernel 2: scores via bf16 WMMA.
// Block = 4 waves, persistent over 32 strips. Ua (bf16, 128 KB) is staged once
// into LDS and B-fragments are read with ds_load_b128; keys stream from HBM
// exactly once (fp32 -> bf16 in regs). Fused epilogue: tanh(D+qpb) dot Va.
// grid: (B*S/16)/(WVS*SPW) = 512 blocks of 128 threads, dynamic LDS 128 KB.
// ---------------------------------------------------------------------------
__global__ __launch_bounds__(128)
void bah_scores(const float* __restrict__ keys, const __bf16* __restrict__ ua_bf,
                const float* __restrict__ qpb,  const float* __restrict__ va,
                float* __restrict__ scores) {
    extern __shared__ __align__(16) char smem_raw[];
    __bf16* uasm = (__bf16*)smem_raw;                 // [HDc][HEc] bf16

    const int tid = threadIdx.x;
    // ---- cooperative Ua copy: global -> LDS, 16B chunks, 64 iters ----
    {
        const vf4* src = (const vf4*)ua_bf;
        vf4*       dst = (vf4*)uasm;
        #pragma unroll
        for (int i = 0; i < (HDc * HEc * 2 / 16) / 128; ++i)
            dst[tid + i * 128] = src[tid + i * 128];
    }
    __syncthreads();

    const int lane = tid & 31;
    const int wave = tid >> 5;
    const int m    = lane & 15;                        // tile row / column n
    const int half = lane >> 4;                        // lane group 0/1

    for (int sp = 0; sp < SPW; ++sp) {
        const int strip = (blockIdx.x * WVS + wave) * SPW + sp;
        const int b     = strip >> 9;                  // strip / (S/16)
        const int s0    = (strip & 511) << 4;

        const float* krow = keys + ((size_t)b * SS + (size_t)(s0 + m)) * HEc;
        if (sp + 1 < SPW)                              // global_prefetch_b8
            __builtin_prefetch(krow + 16 * HEc, 0, 1);

        // ---- A fragments: keys tile, fp32 -> bf16, per 16-bit A layout ----
        v16bf a[8];
        #pragma unroll
        for (int kb = 0; kb < 8; ++kb) {
            const int ka = kb * 32 + half * 8;
            vf4 f0 = *(const vf4*)(krow + ka);
            vf4 f1 = *(const vf4*)(krow + ka + 4);
            vf4 f2 = *(const vf4*)(krow + ka + 16);
            vf4 f3 = *(const vf4*)(krow + ka + 20);
            v16bf av;
            #pragma unroll
            for (int i = 0; i < 4; ++i) {
                av[i]      = (__bf16)f0[i];
                av[4 + i]  = (__bf16)f1[i];
                av[8 + i]  = (__bf16)f2[i];
                av[12 + i] = (__bf16)f3[i];
            }
            a[kb] = av;
        }

        float sacc[8];
        #pragma unroll
        for (int r = 0; r < 8; ++r) sacc[r] = 0.0f;

        // ---- 16 column tiles of HD; 8 WMMAs each (K=256), B from LDS ----
        for (int nt = 0; nt < 16; ++nt) {
            const __bf16* brow = uasm + (size_t)(nt * 16 + m) * HEc;
            v8f c = {0.f, 0.f, 0.f, 0.f, 0.f, 0.f, 0.f, 0.f};
            #pragma unroll
            for (int kb = 0; kb < 8; ++kb) {
                const __bf16* bp = brow + kb * 32 + half * 8;
                v8bf b0 = *(const v8bf*)(bp);          // ds_load_b128
                v8bf b1 = *(const v8bf*)(bp + 16);     // ds_load_b128
                v16bf bv;
                #pragma unroll
                for (int i = 0; i < 8; ++i) { bv[i] = b0[i]; bv[8 + i] = b1[i]; }
                c = __builtin_amdgcn_wmma_f32_16x16x32_bf16(
                        false, a[kb], false, bv, (short)0, c, false, false);
            }
            const float qv  = qpb[b * HDc + nt * 16 + m];
            const float vav = va[nt * 16 + m];
            #pragma unroll
            for (int r = 0; r < 8; ++r)
                sacc[r] = fmaf(fast_tanh(c[r] + qv), vav, sacc[r]);
        }

        // ---- reduce the 16 lanes sharing each row (xor within halves) ----
        #pragma unroll
        for (int r = 0; r < 8; ++r) {
            float v = sacc[r];
            v += __shfl_xor(v, 1, 32);
            v += __shfl_xor(v, 2, 32);
            v += __shfl_xor(v, 4, 32);
            v += __shfl_xor(v, 8, 32);
            sacc[r] = v;
        }
        if (m == 0) {  // lane 0 -> rows 0..7, lane 16 -> rows 8..15
            float* sptr = scores + (size_t)b * SS + s0 + half * 8;
            vf4 lo = {sacc[0], sacc[1], sacc[2], sacc[3]};
            vf4 hi = {sacc[4], sacc[5], sacc[6], sacc[7]};
            *(vf4*)(sptr)     = lo;
            *(vf4*)(sptr + 4) = hi;
        }
    }
}

// ---------------------------------------------------------------------------
// Kernel 3: per-batch softmax stats (max, 1/sum_exp). grid: 32 x 256.
// ---------------------------------------------------------------------------
__global__ __launch_bounds__(256)
void bah_stats(const float* __restrict__ scores, float* __restrict__ stats) {
    const int b = blockIdx.x, tid = threadIdx.x;
    __shared__ float red[256];
    const float* sr = scores + (size_t)b * SS;
    float mx = -INFINITY;
    for (int s = tid; s < SS; s += 256) mx = fmaxf(mx, sr[s]);
    red[tid] = mx; __syncthreads();
    for (int o = 128; o > 0; o >>= 1) {
        if (tid < o) red[tid] = fmaxf(red[tid], red[tid + o]);
        __syncthreads();
    }
    mx = red[0]; __syncthreads();
    float sum = 0.0f;
    for (int s = tid; s < SS; s += 256) sum += __expf(sr[s] - mx);
    red[tid] = sum; __syncthreads();
    for (int o = 128; o > 0; o >>= 1) {
        if (tid < o) red[tid] += red[tid + o];
        __syncthreads();
    }
    if (tid == 0) { stats[b * 2] = mx; stats[b * 2 + 1] = 1.0f / red[0]; }
}

// ---------------------------------------------------------------------------
// Kernel 4: weights out + context (weighted row-sum of keys, bandwidth bound).
// grid: B * (S/256) = 1024 blocks of 256 threads; thread = one channel e.
// ---------------------------------------------------------------------------
__global__ __launch_bounds__(256)
void bah_ctx(const float* __restrict__ scores, const float* __restrict__ stats,
             const float* __restrict__ keys, float* __restrict__ out_ctx,
             float* __restrict__ out_w) {
    const int blk = blockIdx.x;
    const int b = blk >> 5, chunk = blk & 31;
    const int s0 = chunk * 256;
    const int tid = threadIdx.x;
    __shared__ float wsh[256];
    const float mx = stats[b * 2], inv = stats[b * 2 + 1];
    const float w = __expf(scores[(size_t)b * SS + s0 + tid] - mx) * inv;
    wsh[tid] = w;
    out_w[(size_t)b * SS + s0 + tid] = w;
    __syncthreads();
    float acc = 0.0f;
    const float* kb = keys + ((size_t)b * SS + s0) * HEc + tid;
    #pragma unroll 4
    for (int s = 0; s < 256; ++s) acc = fmaf(wsh[s], kb[(size_t)s * HEc], acc);
    atomicAdd(&out_ctx[b * HEc + tid], acc);
}

// ---------------------------------------------------------------------------
extern "C" void kernel_launch(void* const* d_in, const int* in_sizes, int n_in,
                              void* d_out, int out_size, void* d_ws, size_t ws_size,
                              hipStream_t stream) {
    const float* query = (const float*)d_in[0];
    const float* keys  = (const float*)d_in[1];
    const float* Wa_w  = (const float*)d_in[2];
    const float* Wa_b  = (const float*)d_in[3];
    const float* Ua_w  = (const float*)d_in[4];
    const float* Ua_b  = (const float*)d_in[5];
    const float* Va_w  = (const float*)d_in[6];
    // d_in[7] = Va_b: softmax is shift-invariant, weights/context unaffected.

    char* ws = (char*)d_ws;
    __bf16* ua_bf  = (__bf16*)ws;                               // 128 KB
    float*  qpb    = (float*)(ws + 131072);                     // 32 KB
    float*  scores = (float*)(ws + 131072 + 32768);             // 1 MB
    float*  stats  = (float*)(ws + 131072 + 32768 + 1048576);   // 256 B

    float* out_ctx = (float*)d_out;            // [B, 1, HE] first
    float* out_w   = out_ctx + BB * HEc;       // then [B, 1, S]

    hipMemsetAsync(d_out, 0, (size_t)BB * HEc * sizeof(float), stream);

    bah_prep<<<BB, 256, 0, stream>>>(query, Wa_w, Wa_b, Ua_w, Ua_b, ua_bf, qpb);

    const size_t lds_bytes = (size_t)HDc * HEc * sizeof(__bf16);  // 128 KB
    bah_scores<<<(BB * SS / 16) / (WVS * SPW), 128, lds_bytes, stream>>>(
        keys, ua_bf, qpb, Va_w, scores);

    bah_stats<<<BB, 256, 0, stream>>>(scores, stats);
    bah_ctx<<<BB * (SS / 256), 256, 0, stream>>>(scores, stats, keys, out_ctx, out_w);
}
